// OriginalMultiHeadAttention_8839042695281
// MI455X (gfx1250) — matrix operand order repaired and checked by reading, not compiled, not run on hardware
//
#include <hip/hip_runtime.h>
#include <math.h>

// Problem constants (from reference): B=2, S=2048, E=1024, H=16, Dh=64
#define BB 2
#define SS 2048
#define EE 1024
#define HH 16
#define DH 64
#define NWAVES 8   // 256 threads per block = 8 wave32

typedef __attribute__((ext_vector_type(8)))  __bf16 v8bf;
typedef __attribute__((ext_vector_type(16))) __bf16 v16bf;
typedef __attribute__((ext_vector_type(8)))  float  v8f;

// ---------------------------------------------------------------- WMMA helper
__device__ __forceinline__ v8f wmma_bf16(v16bf a, v16bf b, v8f c) {
  // D = A(16x32 bf16) * B(32x16 bf16) + C(16x16 f32)
  return __builtin_amdgcn_wmma_f32_16x16x32_bf16(
      /*neg_a=*/false, a, /*neg_b=*/false, b,
      /*c_mod=*/(short)0, c, /*reuse_a=*/false, /*reuse_b=*/false);
}

// A-fragment (16x32, row-major source, ld in elements).
// Layout used here: lane L -> row M=L&15; elem e: K = 16*half + e  (16 contiguous)
__device__ __forceinline__ v16bf a_frag(const __bf16* p, int ld) {
  const int lane = threadIdx.x & 31;
  const int row  = lane & 15, half = lane >> 4;
  const __bf16* q = p + row * ld + half * 16;
  v8bf lo = *(const v8bf*)(q);
  v8bf hi = *(const v8bf*)(q + 8);
  v16bf a;
#pragma unroll
  for (int i = 0; i < 8; ++i) { a[i] = lo[i]; a[i + 8] = hi[i]; }
  return a;
}

// B-fragment (32x16). Source must be B^T row-major (N rows, K contiguous).
// Layout used here: lane L -> col N=L&15; elem e: K = 16*half + e  (16 contiguous)
__device__ __forceinline__ v16bf b_frag(const __bf16* p, int ld) {
  const int lane = threadIdx.x & 31;
  const int col  = lane & 15, half = lane >> 4;
  return *(const v16bf*)(p + col * ld + half * 16);
}

// -------------------------------------------------- async global->LDS copy
// GLOBAL_LOAD_ASYNC_TO_LDS_B128 (opcode 98): VDST = LDS byte address,
// VADDR = 64-bit global address, tracked by ASYNCcnt. 16B per lane.
__device__ __forceinline__ void async_copy16(const __bf16* g, __bf16* l) {
  asm volatile("global_load_async_to_lds_b128 %0, %1, off"
               :: "v"((unsigned int)(uintptr_t)l),
                  "v"((unsigned long long)(uintptr_t)g)
               : "memory");
}

// ------------------------------------------------------- lane-group reductions
template <int M>
__device__ __forceinline__ float swz_xor(float x) {
  int i = __builtin_bit_cast(int, x);
  int j = __builtin_amdgcn_ds_swizzle(i, (M << 10) | 0x1f); // xor_mask=M, and=0x1f
  return __builtin_bit_cast(float, j);
}
__device__ __forceinline__ float red_max16(float x) {
  x = fmaxf(x, swz_xor<1>(x)); x = fmaxf(x, swz_xor<2>(x));
  x = fmaxf(x, swz_xor<4>(x)); x = fmaxf(x, swz_xor<8>(x));
  return x;
}
__device__ __forceinline__ float red_sum16(float x) {
  x += swz_xor<1>(x); x += swz_xor<2>(x);
  x += swz_xor<4>(x); x += swz_xor<8>(x);
  return x;
}

// ----------------------------------------------------------- convert kernels
__global__ void k_cvt_bf16(const float* __restrict__ in, __bf16* __restrict__ out, int n) {
  int i = blockIdx.x * blockDim.x + threadIdx.x;
  if (i < n) out[i] = (__bf16)in[i];
}
// W[h][e][d] -> WT[h][d][e]  (bf16)
__global__ void k_transpose_w(const float* __restrict__ W, __bf16* __restrict__ WT) {
  int i = blockIdx.x * blockDim.x + threadIdx.x;
  if (i >= HH * EE * DH) return;
  int e = i % EE;
  int d = (i / EE) % DH;
  int h = i / (EE * DH);
  WT[i] = (__bf16)W[(h * EE + e) * DH + d];
}
// Wo[i][o] -> WoT[o][i]  (bf16)
__global__ void k_transpose_wo(const float* __restrict__ W, __bf16* __restrict__ WT) {
  int i = blockIdx.x * blockDim.x + threadIdx.x;
  if (i >= EE * EE) return;
  int in_ = i % EE;
  int o   = i / EE;
  WT[i] = (__bf16)W[in_ * EE + o];
}

// --------------------------------------------------------------- fused QKV
// One wave per 16x16 output tile of one head; X A-fragments shared by q/k/v.
// q: [B,H,S,Dh] bf16 (pre-scaled by 1/sqrt(Dh));  k: [B,H,S,Dh];  v: [B,H,Dh,S] (transposed)
__global__ void k_qkv(const __bf16* __restrict__ xb,
                      const __bf16* __restrict__ WqT, const __bf16* __restrict__ WkT,
                      const __bf16* __restrict__ WvT,
                      const float* __restrict__ bq, const float* __restrict__ bk,
                      const float* __restrict__ bv,
                      __bf16* __restrict__ qb, __bf16* __restrict__ kb,
                      __bf16* __restrict__ vt) {
  const int lane = threadIdx.x & 31;
  const int wave = threadIdx.x >> 5;
  const int tile = blockIdx.x * NWAVES + wave;     // 16384 tiles total
  const int dn = tile & 3;                         // Dh/16
  const int sm = (tile >> 2) & 127;                // S/16
  const int h  = (tile >> 9) & 15;                 // H
  const int b  = (tile >> 13) & 1;                 // B

  const __bf16* A  = xb  + (size_t)(b * SS + sm * 16) * EE;
  const __bf16* Bq = WqT + (size_t)(h * DH + dn * 16) * EE;
  const __bf16* Bk = WkT + (size_t)(h * DH + dn * 16) * EE;
  const __bf16* Bv = WvT + (size_t)(h * DH + dn * 16) * EE;

  v8f cq = {}, ck = {}, cv = {};
  for (int kk = 0; kk < EE; kk += 32) {
    v16bf a = a_frag(A + kk, EE);
    cq = wmma_bf16(a, b_frag(Bq + kk, EE), cq);
    ck = wmma_bf16(a, b_frag(Bk + kk, EE), ck);
    cv = wmma_bf16(a, b_frag(Bv + kk, EE), cv);
  }

  const int col = lane & 15, half = lane >> 4;
  const int d = dn * 16 + col;
  const float biq = bq[h * DH + d], bik = bk[h * DH + d], biv = bv[h * DH + d];
  const size_t bh = (size_t)(b * HH + h);
#pragma unroll
  for (int r = 0; r < 8; ++r) {
    int s = sm * 16 + r + 8 * half;
    qb[(bh * SS + s) * DH + d] = (__bf16)(0.125f * (cq[r] + biq)); // 1/sqrt(64)
    kb[(bh * SS + s) * DH + d] = (__bf16)(ck[r] + bik);
    vt[(bh * DH + d) * SS + s] = (__bf16)(cv[r] + biv);
  }
}

// ------------------------------------------------------- causal flash attention
// Block = 8 waves = 8 consecutive 16-row Q tiles of one (b,h). Key/value tiles
// (32 keys) are cooperatively staged into LDS with async copies, double-buffered;
// per-wave online softmax; P transposed D->A layout through private LDS tile.
__global__ void k_attn(const __bf16* __restrict__ qb, const __bf16* __restrict__ kb,
                       const __bf16* __restrict__ vt, __bf16* __restrict__ concat) {
  const int tid  = threadIdx.x;
  const int lane = tid & 31;
  const int wave = tid >> 5;

  __shared__ __align__(32) __bf16 ldsK[2][32 * DH];   // K tile  [t][d]   4KB x2
  __shared__ __align__(32) __bf16 ldsV[2][DH * 32];   // V^T tile [d][t]  4KB x2
  __shared__ __align__(32) __bf16 Plds[NWAVES][16 * 32]; // per-wave P tile 8KB
  __bf16* pl = &Plds[wave][0];

  const int bid = blockIdx.x;              // 512 blocks
  const int qb7 = bid & 15;                // 16 blocks of 128 rows per (b,h)
  const int h   = (bid >> 4) & 15;
  const int b   = (bid >> 8) & 1;
  const size_t bh = (size_t)(b * HH + h);
  const int blockQ0 = qb7 * 128;
  const int qRow0   = blockQ0 + wave * 16;

  const __bf16* Qb = qb + (bh * SS + qRow0) * DH;
  const __bf16* Kb = kb + bh * SS * DH;
  const __bf16* Vb = vt + bh * DH * SS;

  // Q A-fragments stay resident (Kdim 0..31, 32..63)
  v16bf aq0 = a_frag(Qb + 0, DH);
  v16bf aq1 = a_frag(Qb + 32, DH);

  const int col = lane & 15, half = lane >> 4;
  const int rowb = qRow0 + 8 * half;

  float m[8], l[8];
  v8f acc[4] = {};
#pragma unroll
  for (int r = 0; r < 8; ++r) { m[r] = -3.0e38f; l[r] = 0.f; }

  // cooperative async tile fetch: 256 threads x 16B = 4KB per tile
  const int krow = tid >> 3, kc8 = tid & 7;   // K: row 0..31, 16B-chunk 0..7
  const int vrow = tid >> 2, vc8 = tid & 3;   // V: d  0..63, chunk 0..3
  auto issue_tile = [&](int t0, int buf) {
    async_copy16(Kb + (size_t)(t0 + krow) * DH + kc8 * 8,
                 &ldsK[buf][krow * DH + kc8 * 8]);
    async_copy16(Vb + (size_t)vrow * SS + t0 + vc8 * 8,
                 &ldsV[buf][vrow * 32 + vc8 * 8]);
  };

  const int nIter = blockQ0 / 32 + 4;       // t0 = i*32 covers keys <= blockQ0+127
  issue_tile(0, 0);
  if (nIter > 1) issue_tile(32, 1);

  for (int i = 0; i < nIter; ++i) {
    const int t0  = i * 32;
    const int buf = i & 1;
    // loads complete in order: <=2 outstanding => this tile's 2 ops are done
    if (i + 1 < nIter) asm volatile("s_wait_asynccnt 0x2" ::: "memory");
    else               asm volatile("s_wait_asynccnt 0x0" ::: "memory");
    __syncthreads();                         // tile visible to all waves

    if (t0 <= qRow0 + 15) {                  // wave-uniform causal extent
      const __bf16* Kt = &ldsK[buf][0];
      const __bf16* Vt = &ldsV[buf][0];
      v8f s0 = {}, s1 = {};
      s0 = wmma_bf16(aq0, b_frag(Kt + 0 * DH + 0,  DH), s0);
      s0 = wmma_bf16(aq1, b_frag(Kt + 0 * DH + 32, DH), s0);
      s1 = wmma_bf16(aq0, b_frag(Kt + 16 * DH + 0,  DH), s1);
      s1 = wmma_bf16(aq1, b_frag(Kt + 16 * DH + 32, DH), s1);

      float p0[8], p1[8];
#pragma unroll
      for (int r = 0; r < 8; ++r) {
        const int row = rowb + r;
        float v0 = (t0 + col      > row) ? -3.0e38f : s0[r];   // causal mask
        float v1 = (t0 + 16 + col > row) ? -3.0e38f : s1[r];
        float tm = red_max16(fmaxf(v0, v1));                   // row max (16 lanes)
        float mn = fmaxf(m[r], tm);
        p0[r] = __expf(v0 - mn);
        p1[r] = __expf(v1 - mn);
        float ts = red_sum16(p0[r] + p1[r]);
        float alpha = __expf(m[r] - mn);
        l[r] = l[r] * alpha + ts;
        m[r] = mn;
#pragma unroll
        for (int dt = 0; dt < 4; ++dt) acc[dt][r] *= alpha;
      }

      // D-layout -> A-layout transpose of P via per-wave LDS tile
#pragma unroll
      for (int r = 0; r < 8; ++r) {
        const int row = r + 8 * half;
        pl[row * 32 + col]      = (__bf16)p0[r];
        pl[row * 32 + 16 + col] = (__bf16)p1[r];
      }
      asm volatile("s_wait_dscnt 0" ::: "memory"); // RAW: ds stores -> ds loads
      v16bf ap = a_frag(pl, 32);

#pragma unroll
      for (int dt = 0; dt < 4; ++dt)
        acc[dt] = wmma_bf16(ap, b_frag(Vt + dt * 16 * 32, 32), acc[dt]);
    }

    __syncthreads();                         // everyone done reading buf
    if (i + 2 < nIter) issue_tile(t0 + 64, buf);
  }

  // normalize and write concat [B,S,H*Dh] as bf16
#pragma unroll
  for (int r = 0; r < 8; ++r) {
    const float inv = 1.0f / l[r];
    const int s = qRow0 + r + 8 * half;
#pragma unroll
    for (int dt = 0; dt < 4; ++dt)
      concat[((size_t)(b * SS + s)) * EE + h * DH + dt * 16 + col] =
          (__bf16)(acc[dt][r] * inv);
  }
}

// ------------------------------------------------------------- output proj
__global__ void k_oproj(const __bf16* __restrict__ X, const __bf16* __restrict__ WoT,
                        const float* __restrict__ bo, float* __restrict__ out) {
  const int lane = threadIdx.x & 31;
  const int wave = threadIdx.x >> 5;
  const int tile = blockIdx.x * NWAVES + wave;  // 256 * 64 tiles
  const int nt = tile & 63;
  const int mt = tile >> 6;

  const __bf16* A  = X   + (size_t)mt * 16 * EE;
  const __bf16* Bp = WoT + (size_t)nt * 16 * EE;

  v8f c = {};
  for (int kk = 0; kk < EE; kk += 32)
    c = wmma_bf16(a_frag(A + kk, EE), b_frag(Bp + kk, EE), c);

  const int col = lane & 15, half = lane >> 4;
  const float bias = bo[nt * 16 + col];
#pragma unroll
  for (int r = 0; r < 8; ++r)
    out[((size_t)(mt * 16 + r + 8 * half)) * EE + nt * 16 + col] = c[r] + bias;
}

// ---------------------------------------------------------------- launcher
extern "C" void kernel_launch(void* const* d_in, const int* in_sizes, int n_in,
                              void* d_out, int out_size, void* d_ws, size_t ws_size,
                              hipStream_t stream) {
  const float* x  = (const float*)d_in[0];
  const float* Wq = (const float*)d_in[1];
  const float* bq = (const float*)d_in[2];
  const float* Wk = (const float*)d_in[3];
  const float* bk = (const float*)d_in[4];
  const float* Wv = (const float*)d_in[5];
  const float* bv = (const float*)d_in[6];
  const float* Wo = (const float*)d_in[7];
  const float* bo = (const float*)d_in[8];

  // workspace layout (48 MB total, MB-aligned segments)
  char* ws = (char*)d_ws;
  __bf16* xb   = (__bf16*)(ws + ((size_t)0  << 20)); //  8 MB: x as bf16
  __bf16* WqT  = (__bf16*)(ws + ((size_t)8  << 20)); //  2 MB
  __bf16* WkT  = (__bf16*)(ws + ((size_t)10 << 20)); //  2 MB
  __bf16* WvT  = (__bf16*)(ws + ((size_t)12 << 20)); //  2 MB
  __bf16* WoT  = (__bf16*)(ws + ((size_t)14 << 20)); //  2 MB
  __bf16* qbuf = (__bf16*)(ws + ((size_t)16 << 20)); //  8 MB
  __bf16* kbuf = (__bf16*)(ws + ((size_t)24 << 20)); //  8 MB
  __bf16* vtb  = (__bf16*)(ws + ((size_t)32 << 20)); //  8 MB  (V transposed)
  __bf16* cat  = (__bf16*)(ws + ((size_t)40 << 20)); //  8 MB

  const int nX = BB * SS * EE;        // 4 Mi
  const int nW = HH * EE * DH;        // 1 Mi
  k_cvt_bf16<<<(nX + 255) / 256, 256, 0, stream>>>(x, xb, nX);
  k_transpose_w<<<(nW + 255) / 256, 256, 0, stream>>>(Wq, WqT);
  k_transpose_w<<<(nW + 255) / 256, 256, 0, stream>>>(Wk, WkT);
  k_transpose_w<<<(nW + 255) / 256, 256, 0, stream>>>(Wv, WvT);
  k_transpose_wo<<<(EE * EE + 255) / 256, 256, 0, stream>>>(Wo, WoT);

  // 16384 tiles / 8 waves = 2048 blocks
  k_qkv<<<2048, NWAVES * 32, 0, stream>>>(xb, WqT, WkT, WvT, bq, bk, bv,
                                          qbuf, kbuf, vtb);
  // 8 q-tiles per block, 128 q-tiles per (b,h), B*H = 32 -> 512 blocks
  k_attn<<<512, NWAVES * 32, 0, stream>>>(qbuf, kbuf, vtb, cat);
  // 16384 tiles / 8 = 2048 blocks
  k_oproj<<<2048, NWAVES * 32, 0, stream>>>(cat, WoT, bo, (float*)d_out);
}
